// IMM_block_71339406786841
// MI455X (gfx1250) — matrix operand order repaired and checked by reading, not compiled, not run on hardware
//
#include <hip/hip_runtime.h>
#include <hip/hip_bf16.h>
#include <stdint.h>

// ---------------------------------------------------------------------------
// MI455X (gfx1250) masked bi-Mamba block.
//
//  * Dense GEMMs -> bf16 WMMA (v_wmma_f32_16x16x32_bf16, f32 accum), block
//    tile 64x64, 8 waves, 2 WMMA per wave per K-step, LDS-staged with
//    b128 loads/stores and register double-buffering of global loads.
//  * Selective scan -> 1024-way channel-parallel serial recurrence.
//  * wave32 WMMA fragment layouts per CDNA5 ISA 7.12.2:
//      A lane(g*16+r): row r, K(e) = (e&7) + (e>=8?16:0) + 8g  (two 8-runs)
//      B lane(g*16+r): col r, K(e) = same as A               (two 8-runs;
//        so B is staged TRANSPOSED in LDS: Bst[N][K])
//      D vgpr v      : row 8g+v, col r
// ---------------------------------------------------------------------------

typedef __bf16 bf16;
typedef bf16  v16bf __attribute__((ext_vector_type(16)));
typedef bf16  v8bf  __attribute__((ext_vector_type(8)));
typedef bf16  v2bf  __attribute__((ext_vector_type(2)));
typedef float v8f   __attribute__((ext_vector_type(8)));

#define D_MODEL 512
#define D_INNER 1024
#define D_STATE 16
#define DT_RANK 32
#define SEQ_LEN 4096
#define K_MASK  819                 // int(0.2 * 4096)
#define L_KEEP  (SEQ_LEN - K_MASK)  // 3277

#define GEMM_TM 64
#define GEMM_TN 64
#define GEMM_TK 32
#define LDS_PITCH 40                // bf16 elements; 80B rows -> 16B aligned runs

static __device__ __forceinline__ float siluf(float v) {
    return v / (1.f + __expf(-v));
}

// ---------------------------------------------------------------------------
// fp32 -> bf16 weight conversion, 2 elements/thread (packed cvt + b32 store)
// ---------------------------------------------------------------------------
__global__ void k_cvt_bf16(const float* __restrict__ s, bf16* __restrict__ d, int npairs) {
    int i = blockIdx.x * blockDim.x + threadIdx.x;
    if (i >= npairs) return;
    float2 v = ((const float2*)s)[i];
    v2bf o;
    o[0] = (bf16)v.x;
    o[1] = (bf16)v.y;
    ((v2bf*)d)[i] = o;
}

// ---------------------------------------------------------------------------
// C[M x N] = A[M x K](f32, row stride lda) * B[K x N](bf16, row-major)
// Block: 256 threads = 8 waves, tile 64x64, K step 32.
// N % 64 == 0, K % 32 == 0 (true for all shapes used); M guarded by clamping
// (over-read rows feed accumulators whose stores are skipped).
// ---------------------------------------------------------------------------
__global__ __launch_bounds__(256) void k_gemm_a32_bbf16(
    const float* __restrict__ A, int lda,
    const bf16* __restrict__ B,
    float* __restrict__ C,
    int M, int N, int K)
{
    __shared__ __attribute__((aligned(16))) bf16 As [GEMM_TM][LDS_PITCH]; // [m][k]
    __shared__ __attribute__((aligned(16))) bf16 Bst[GEMM_TN][LDS_PITCH]; // [n][k] (transposed)

    const int t    = threadIdx.x;
    const int bm   = blockIdx.y * GEMM_TM;
    const int bn   = blockIdx.x * GEMM_TN;
    const int lane = t & 31;
    const int wv   = t >> 5;
    const int wrow = wv >> 2;          // 0..1
    const int wcol = wv & 3;           // 0..3
    const int r    = lane & 15;
    const int g    = lane >> 4;

    // Staging assignments: 8 contiguous elements per thread.
    const int tm  = (t * 8) >> 5;      // A tile row   0..63
    const int tk  = (t * 8) & 31;      // A tile col   {0,8,16,24}
    const int bk  = (t * 8) >> 6;      // B tile k     0..31
    const int bn0 = (t * 8) & 63;      // B tile n     {0,8,...,56}

    int arow = bm + tm; if (arow >= M) arow = M - 1;   // clamp: branch-free staging
    const float* Arow = A + (size_t)arow * lda;

    v8f acc0 = {0.f,0.f,0.f,0.f,0.f,0.f,0.f,0.f};
    v8f acc1 = {0.f,0.f,0.f,0.f,0.f,0.f,0.f,0.f};

    // Register double buffer: global loads for panel k0 issued one step early.
    float4 ra0, ra1;
    uint4  rb;
    {
        const float4* ap = (const float4*)(Arow + tk);
        ra0 = ap[0]; ra1 = ap[1];
        rb  = *(const uint4*)(B + (size_t)bk * N + bn + bn0);
    }

    for (int k0 = 0; k0 < K; k0 += GEMM_TK) {
        // ---- stage registers -> LDS (bf16) ----
        {
            v8bf av;
            av[0] = (bf16)ra0.x; av[1] = (bf16)ra0.y;
            av[2] = (bf16)ra0.z; av[3] = (bf16)ra0.w;
            av[4] = (bf16)ra1.x; av[5] = (bf16)ra1.y;
            av[6] = (bf16)ra1.z; av[7] = (bf16)ra1.w;
            *(v8bf*)&As[tm][tk] = av;                      // ds_store_b128

            union { uint4 u; bf16 h[8]; } ub; ub.u = rb;
            #pragma unroll
            for (int j = 0; j < 8; ++j)                    // transposed scatter
                Bst[bn0 + j][bk] = ub.h[j];
        }
        __syncthreads();

        // ---- issue next panel's global loads (overlap with WMMA work) ----
        int kn = k0 + GEMM_TK;
        if (kn < K) {
            const float4* ap = (const float4*)(Arow + kn + tk);
            ra0 = ap[0]; ra1 = ap[1];
            rb  = *(const uint4*)(B + (size_t)(kn + bk) * N + bn + bn0);
            if (kn + GEMM_TK < K)
                __builtin_prefetch(Arow + kn + GEMM_TK + tk, 0, 0);
        }

        // ---- fragments (b128 LDS loads) + 2x WMMA ----
        const int mt0 = wrow * 16;
        const int mt1 = mt0 + 32;
        v8bf alo0 = *(const v8bf*)&As[mt0 + r][g * 8];
        v8bf ahi0 = *(const v8bf*)&As[mt0 + r][g * 8 + 16];
        v8bf alo1 = *(const v8bf*)&As[mt1 + r][g * 8];
        v8bf ahi1 = *(const v8bf*)&As[mt1 + r][g * 8 + 16];
        v8bf blo  = *(const v8bf*)&Bst[wcol * 16 + r][g * 8];
        v8bf bhi  = *(const v8bf*)&Bst[wcol * 16 + r][g * 8 + 16];

        v16bf af0 = __builtin_shufflevector(alo0, ahi0, 0,1,2,3,4,5,6,7,8,9,10,11,12,13,14,15);
        v16bf af1 = __builtin_shufflevector(alo1, ahi1, 0,1,2,3,4,5,6,7,8,9,10,11,12,13,14,15);
        v16bf bf_ = __builtin_shufflevector(blo,  bhi,  0,1,2,3,4,5,6,7,8,9,10,11,12,13,14,15);

        acc0 = __builtin_amdgcn_wmma_f32_16x16x32_bf16(
            false, af0, false, bf_, (short)0, acc0, false, false);
        acc1 = __builtin_amdgcn_wmma_f32_16x16x32_bf16(
            false, af1, false, bf_, (short)0, acc1, false, false);
        __syncthreads();
    }

    // ---- epilogue: D vgpr v -> row 8g+v, col r ----
    const int col = bn + wcol * 16 + r;
    #pragma unroll
    for (int v = 0; v < 8; ++v) {
        int row0 = bm + wrow * 16 + g * 8 + v;
        if (row0 < M) C[(size_t)row0 * N + col] = acc0[v];
        int row1 = row0 + 32;
        if (row1 < M) C[(size_t)row1 * N + col] = acc1[v];
    }
}

// ---------------------------------------------------------------------------
// LayerNorm over D_MODEL=512, one row per block (256 threads, 2 elems/thread)
// ---------------------------------------------------------------------------
__global__ __launch_bounds__(256) void k_layernorm(
    const float* __restrict__ x, const float* __restrict__ w,
    const float* __restrict__ b, float* __restrict__ y)
{
    __shared__ float s1[256], s2[256];
    int row = blockIdx.x, t = threadIdx.x;
    const float* xr = x + (size_t)row * D_MODEL;
    float a0 = xr[t], a1 = xr[t + 256];
    s1[t] = a0 + a1;
    s2[t] = a0 * a0 + a1 * a1;
    __syncthreads();
    for (int st = 128; st > 0; st >>= 1) {
        if (t < st) { s1[t] += s1[t + st]; s2[t] += s2[t + st]; }
        __syncthreads();
    }
    float mu  = s1[0] * (1.f / D_MODEL);
    float var = s2[0] * (1.f / D_MODEL) - mu * mu;
    float inv = rsqrtf(var + 1e-5f);
    float* yr = y + (size_t)row * D_MODEL;
    yr[t]       = (a0 - mu) * inv * w[t]       + b[t];
    yr[t + 256] = (a1 - mu) * inv * w[t + 256] + b[t + 256];
}

// ---------------------------------------------------------------------------
// Elementwise helpers
// ---------------------------------------------------------------------------
__global__ void k_copy(const float* __restrict__ s, float* __restrict__ d, int n) {
    int i = blockIdx.x * blockDim.x + threadIdx.x;
    if (i < n) d[i] = s[i];
}

__global__ void k_flip_rows(const float* __restrict__ s, float* __restrict__ d, int L) {
    int i = blockIdx.x * blockDim.x + threadIdx.x;
    if (i >= L * D_MODEL) return;
    int l = i / D_MODEL, c = i % D_MODEL;
    d[i] = s[(size_t)(L - 1 - l) * D_MODEL + c];
}

__global__ void k_add_rows(float* __restrict__ dst, const float* __restrict__ src,
                           int L, int flip) {
    int i = blockIdx.x * blockDim.x + threadIdx.x;
    if (i >= L * D_MODEL) return;
    int l = i / D_MODEL, c = i % D_MODEL;
    int ls = flip ? (L - 1 - l) : l;
    dst[i] += src[(size_t)ls * D_MODEL + c];
}

// Depthwise causal conv (width 4) + SiLU; xi = left half of xz (L x 2048)
__global__ void k_conv_silu(const float* __restrict__ xz, const float* __restrict__ cw,
                            const float* __restrict__ cb, float* __restrict__ xc, int L)
{
    int i = blockIdx.x * blockDim.x + threadIdx.x;
    if (i >= L * D_INNER) return;
    int l = i >> 10, d = i & (D_INNER - 1);
    float acc = cb[d];
    float4 w4 = *(const float4*)(cw + d * 4);
    const float wj[4] = {w4.x, w4.y, w4.z, w4.w};
    #pragma unroll
    for (int j = 0; j < 4; ++j) {
        int ls = l - 3 + j;
        if (ls >= 0) acc += wj[j] * xz[(size_t)ls * (2 * D_INNER) + d];
    }
    xc[i] = siluf(acc);
}

// dt = softplus(dt_mm + b_dt[d]) (in place)
__global__ void k_dt_softplus(float* __restrict__ dtb, const float* __restrict__ b_dt, int L)
{
    int i = blockIdx.x * blockDim.x + threadIdx.x;
    if (i >= L * D_INNER) return;
    int d = i & (D_INNER - 1);
    float v = dtb[i] + b_dt[d];
    dtb[i] = (v > 20.f) ? v : log1pf(__expf(v));
}

// Selective scan: 1024 independent channels, serial over L, 16-state each.
__global__ __launch_bounds__(256) void k_scan(
    const float* __restrict__ dt, const float* __restrict__ dbc,
    const float* __restrict__ xc, const float* __restrict__ A_log,
    float* __restrict__ ys, int L)
{
    int d = blockIdx.x * 256 + threadIdx.x;   // 0..1023
    float a[D_STATE], h[D_STATE];
    #pragma unroll
    for (int s = 0; s < D_STATE; ++s) { a[s] = -__expf(A_log[d * D_STATE + s]); h[s] = 0.f; }
    for (int l = 0; l < L; ++l) {
        float dtv = dt[(size_t)l * D_INNER + d];
        float xv  = xc[(size_t)l * D_INNER + d];
        // dbc row: [0:32)=dt_raw [32:48)=B [48:64)=C ; vector loads
        const float4* bc4 = (const float4*)(dbc + (size_t)l * 64);
        float bs[D_STATE], cs[D_STATE];
        #pragma unroll
        for (int q = 0; q < 4; ++q) {
            float4 bv = bc4[8 + q], cv = bc4[12 + q];
            bs[4*q+0] = bv.x; bs[4*q+1] = bv.y; bs[4*q+2] = bv.z; bs[4*q+3] = bv.w;
            cs[4*q+0] = cv.x; cs[4*q+1] = cv.y; cs[4*q+2] = cv.z; cs[4*q+3] = cv.w;
        }
        float dx = dtv * xv, y = 0.f;
        #pragma unroll
        for (int s = 0; s < D_STATE; ++s) {
            h[s] = __expf(dtv * a[s]) * h[s] + dx * bs[s];
            y += h[s] * cs[s];
        }
        ys[(size_t)l * D_INNER + d] = y;
    }
}

// u = (ys + xc*D) * silu(z);  z = right half of xz
__global__ void k_gate(const float* __restrict__ ys, const float* __restrict__ xc,
                       const float* __restrict__ Dp, const float* __restrict__ xz,
                       float* __restrict__ u, int L)
{
    int i = blockIdx.x * blockDim.x + threadIdx.x;
    if (i >= L * D_INNER) return;
    int l = i >> 10, d = i & (D_INNER - 1);
    float z = xz[(size_t)l * (2 * D_INNER) + D_INNER + d];
    u[i] = (ys[i] + xc[i] * Dp[d]) * siluf(z);
}

// logits[row] = tanh(t1[row] + b1) . w2 + b2   (t1 = h @ w1 from WMMA GEMM)
__global__ __launch_bounds__(128) void k_tanh_w2(
    const float* __restrict__ t1, const float* __restrict__ b1,
    const float* __restrict__ w2, const float* __restrict__ b2,
    float* __restrict__ outv)
{
    __shared__ float s[128];
    int row = blockIdx.x, t = threadIdx.x;
    s[t] = tanhf(t1[(size_t)row * 128 + t] + b1[t]) * w2[t];
    __syncthreads();
    for (int st = 64; st > 0; st >>= 1) {
        if (t < st) s[t] += s[t + st];
        __syncthreads();
    }
    if (t == 0) outv[row] = s[0] + b2[0];
}

// Stable rank (softmax is monotone, so rank raw logits): kept if rank < keepN
__global__ void k_rank(const float* __restrict__ sc, int* __restrict__ kept,
                       int L, int keepN)
{
    int i = blockIdx.x * blockDim.x + threadIdx.x;
    if (i >= L) return;
    float si = sc[i];
    int rank = 0;
    for (int j = 0; j < L; ++j) {
        float sj = sc[j];
        rank += (sj > si) || (sj == si && j < i);
    }
    kept[i] = (rank < keepN) ? 1 : 0;
}

// Order-preserving compaction of kept indices (single block)
__global__ __launch_bounds__(1024) void k_compact(
    const int* __restrict__ kept, int* __restrict__ keep, int L)
{
    for (int i = threadIdx.x; i < L; i += 1024) {
        if (kept[i]) {
            int pos = 0;
            for (int j = 0; j < i; ++j) pos += kept[j];
            keep[pos] = i;
        }
    }
}

__global__ void k_gather(const float* __restrict__ h1, const int* __restrict__ keep,
                         float* __restrict__ f, int L2)
{
    int i = blockIdx.x * blockDim.x + threadIdx.x;
    if (i >= L2 * D_MODEL) return;
    int r = i >> 9, c = i & (D_MODEL - 1);
    f[i] = h1[(size_t)keep[r] * D_MODEL + c];
}

// In-place softmax over L values (single block)
__global__ __launch_bounds__(256) void k_softmax_1b(float* __restrict__ sc, int L)
{
    __shared__ float s[256];
    int t = threadIdx.x;
    float m = -1e30f;
    for (int i = t; i < L; i += 256) m = fmaxf(m, sc[i]);
    s[t] = m; __syncthreads();
    for (int st = 128; st > 0; st >>= 1) {
        if (t < st) s[t] = fmaxf(s[t], s[t + st]);
        __syncthreads();
    }
    m = s[0];
    __syncthreads();
    float sum = 0.f;
    for (int i = t; i < L; i += 256) { float e = __expf(sc[i] - m); sc[i] = e; sum += e; }
    s[t] = sum; __syncthreads();
    for (int st = 128; st > 0; st >>= 1) {
        if (t < st) s[t] += s[t + st];
        __syncthreads();
    }
    float inv = 1.f / s[0];
    __syncthreads();
    for (int i = t; i < L; i += 256) sc[i] *= inv;
}

// out[d] = sum_l a2[l] * r2[l][d]
__global__ __launch_bounds__(256) void k_wsum(
    const float* __restrict__ r2, const float* __restrict__ a2,
    float* __restrict__ out, int L)
{
    int d = blockIdx.x * 256 + threadIdx.x;
    if (d >= D_MODEL) return;
    float acc = 0.f;
    for (int l = 0; l < L; ++l) acc += a2[l] * r2[(size_t)l * D_MODEL + d];
    out[d] = acc;
}

// ---------------------------------------------------------------------------
// Host-side orchestration
// ---------------------------------------------------------------------------
struct MambaP {
    const float *W_in, *conv_w, *conv_b, *W_x, *W_dt, *b_dt, *A_log, *Dp, *W_out;
    bf16 *W_in_bf, *W_x_bf, *W_dt_bf, *W_out_bf;
};

static inline int cdiv(int a, int b) { return (a + b - 1) / b; }

extern "C" void kernel_launch(void* const* d_in, const int* in_sizes, int n_in,
                              void* d_out, int out_size, void* d_ws, size_t ws_size,
                              hipStream_t stream)
{
    (void)in_sizes; (void)n_in; (void)out_size; (void)ws_size;

    // Input flattening assumption: setup_inputs() insertion order, nested dicts
    // expanded in their own insertion order:
    //  0:x 1:ln1_w 2:ln1_b 3:norm_w 4:norm_b 5:ln2_w 6:ln2_b
    //  7..15:m1f{W_in,conv_w,conv_b,W_x,W_dt,b_dt,A_log,D,W_out}
    //  16..24:m1b 25..33:m2f 34..42:m2b
    //  43..46:mask_att{w1,b1,w2,b2} 47..50:grp_att{w1,b1,w2,b2}
    const float* x      = (const float*)d_in[0];
    const float* ln1_w  = (const float*)d_in[1];
    const float* ln1_b  = (const float*)d_in[2];
    const float* norm_w = (const float*)d_in[3];
    const float* norm_b = (const float*)d_in[4];
    const float* ln2_w  = (const float*)d_in[5];
    const float* ln2_b  = (const float*)d_in[6];

    // Workspace bump allocator (256B aligned)
    uint8_t* base = (uint8_t*)d_ws;
    size_t off = 0;
    auto alloc_f  = [&](size_t n) -> float* { float* p = (float*)(base + off); off += ((n * 4 + 255) / 256) * 256; return p; };
    auto alloc_bf = [&](size_t n) -> bf16*  { bf16*  p = (bf16*) (base + off); off += ((n * 2 + 255) / 256) * 256; return p; };
    auto alloc_i  = [&](size_t n) -> int*   { int*   p = (int*)  (base + off); off += ((n * 4 + 255) / 256) * 256; return p; };

    auto cvt = [&](const float* s, bf16* d, int n) {
        int np = n / 2;
        k_cvt_bf16<<<cdiv(np, 256), 256, 0, stream>>>(s, d, np);
    };

    auto make_mamba = [&](int b) -> MambaP {
        MambaP p;
        p.W_in   = (const float*)d_in[b + 0];
        p.conv_w = (const float*)d_in[b + 1];
        p.conv_b = (const float*)d_in[b + 2];
        p.W_x    = (const float*)d_in[b + 3];
        p.W_dt   = (const float*)d_in[b + 4];
        p.b_dt   = (const float*)d_in[b + 5];
        p.A_log  = (const float*)d_in[b + 6];
        p.Dp     = (const float*)d_in[b + 7];
        p.W_out  = (const float*)d_in[b + 8];
        p.W_in_bf  = alloc_bf((size_t)D_MODEL * 2 * D_INNER);
        p.W_x_bf   = alloc_bf((size_t)D_INNER * (DT_RANK + 2 * D_STATE));
        p.W_dt_bf  = alloc_bf((size_t)DT_RANK * D_INNER);
        p.W_out_bf = alloc_bf((size_t)D_INNER * D_MODEL);
        cvt(p.W_in,  p.W_in_bf,  D_MODEL * 2 * D_INNER);
        cvt(p.W_x,   p.W_x_bf,   D_INNER * (DT_RANK + 2 * D_STATE));
        cvt(p.W_dt,  p.W_dt_bf,  DT_RANK * D_INNER);
        cvt(p.W_out, p.W_out_bf, D_INNER * D_MODEL);
        return p;
    };

    MambaP m1f = make_mamba(7),  m1b = make_mamba(16);
    MambaP m2f = make_mamba(25), m2b = make_mamba(34);

    const float* mask_w1 = (const float*)d_in[43];
    const float* mask_b1 = (const float*)d_in[44];
    const float* mask_w2 = (const float*)d_in[45];
    const float* mask_b2 = (const float*)d_in[46];
    const float* grp_w1  = (const float*)d_in[47];
    const float* grp_b1  = (const float*)d_in[48];
    const float* grp_w2  = (const float*)d_in[49];
    const float* grp_b2  = (const float*)d_in[50];

    bf16* mask_w1_bf = alloc_bf((size_t)D_MODEL * 128);
    bf16* grp_w1_bf  = alloc_bf((size_t)D_MODEL * 128);
    cvt(mask_w1, mask_w1_bf, D_MODEL * 128);
    cvt(grp_w1,  grp_w1_bf,  D_MODEL * 128);

    // Activation buffers (sized for L = SEQ_LEN, reused by both layers)
    float* xln  = alloc_f((size_t)SEQ_LEN * D_MODEL);
    float* xlnf = alloc_f((size_t)SEQ_LEN * D_MODEL);
    float* xz   = alloc_f((size_t)SEQ_LEN * 2 * D_INNER);
    float* xc   = alloc_f((size_t)SEQ_LEN * D_INNER);
    float* dbc  = alloc_f((size_t)SEQ_LEN * 64);
    float* dtb  = alloc_f((size_t)SEQ_LEN * D_INNER);
    float* ys   = alloc_f((size_t)SEQ_LEN * D_INNER);
    float* u    = alloc_f((size_t)SEQ_LEN * D_INNER);
    float* mo   = alloc_f((size_t)SEQ_LEN * D_MODEL);
    float* h1   = alloc_f((size_t)SEQ_LEN * D_MODEL);
    float* hn   = alloc_f((size_t)SEQ_LEN * D_MODEL);
    float* t1   = alloc_f((size_t)SEQ_LEN * 128);
    float* sc   = alloc_f(SEQ_LEN);
    int*   kept = alloc_i(SEQ_LEN);
    int*   keep = alloc_i(L_KEEP);
    float* fbuf = alloc_f((size_t)L_KEEP * D_MODEL);
    float* f1   = alloc_f((size_t)L_KEEP * D_MODEL);

    auto gemm = [&](const float* A, int lda, const bf16* B, float* C,
                    int M, int N, int K) {
        dim3 g(N / GEMM_TN, cdiv(M, GEMM_TM));
        k_gemm_a32_bbf16<<<g, 256, 0, stream>>>(A, lda, B, C, M, N, K);
    };

    auto run_mamba = [&](const float* xin, const MambaP& p, float* out_md, int L) {
        gemm(xin, D_MODEL, p.W_in_bf, xz, L, 2 * D_INNER, D_MODEL);
        k_conv_silu<<<cdiv(L * D_INNER, 256), 256, 0, stream>>>(xz, p.conv_w, p.conv_b, xc, L);
        gemm(xc, D_INNER, p.W_x_bf, dbc, L, 64, D_INNER);
        gemm(dbc, 64, p.W_dt_bf, dtb, L, D_INNER, DT_RANK);
        k_dt_softplus<<<cdiv(L * D_INNER, 256), 256, 0, stream>>>(dtb, p.b_dt, L);
        k_scan<<<4, 256, 0, stream>>>(dtb, dbc, xc, p.A_log, ys, L);
        k_gate<<<cdiv(L * D_INNER, 256), 256, 0, stream>>>(ys, xc, p.Dp, xz, u, L);
        gemm(u, D_INNER, p.W_out_bf, out_md, L, D_MODEL, D_INNER);
    };

    // -------- Layer 1: h1 = bimamba(LN1(x)) + x --------
    k_layernorm<<<SEQ_LEN, 256, 0, stream>>>(x, ln1_w, ln1_b, xln);
    k_flip_rows<<<cdiv(SEQ_LEN * D_MODEL, 256), 256, 0, stream>>>(xln, xlnf, SEQ_LEN);
    k_copy<<<cdiv(SEQ_LEN * D_MODEL, 256), 256, 0, stream>>>(x, h1, SEQ_LEN * D_MODEL);
    run_mamba(xln, m1f, mo, SEQ_LEN);
    k_add_rows<<<cdiv(SEQ_LEN * D_MODEL, 256), 256, 0, stream>>>(h1, mo, SEQ_LEN, 0);
    run_mamba(xlnf, m1b, mo, SEQ_LEN);
    k_add_rows<<<cdiv(SEQ_LEN * D_MODEL, 256), 256, 0, stream>>>(h1, mo, SEQ_LEN, 1);

    // -------- Mask scoring + top-(L-k) selection --------
    k_layernorm<<<SEQ_LEN, 256, 0, stream>>>(h1, norm_w, norm_b, hn);
    gemm(hn, D_MODEL, mask_w1_bf, t1, SEQ_LEN, 128, D_MODEL);
    k_tanh_w2<<<SEQ_LEN, 128, 0, stream>>>(t1, mask_b1, mask_w2, mask_b2, sc);
    k_rank<<<cdiv(SEQ_LEN, 256), 256, 0, stream>>>(sc, kept, SEQ_LEN, L_KEEP);
    k_compact<<<1, 1024, 0, stream>>>(kept, keep, SEQ_LEN);
    k_gather<<<cdiv(L_KEEP * D_MODEL, 256), 256, 0, stream>>>(h1, keep, fbuf, L_KEEP);

    // -------- Layer 2: f1 = bimamba(LN2(f)) + f --------
    k_layernorm<<<L_KEEP, 256, 0, stream>>>(fbuf, ln2_w, ln2_b, xln);
    k_flip_rows<<<cdiv(L_KEEP * D_MODEL, 256), 256, 0, stream>>>(xln, xlnf, L_KEEP);
    k_copy<<<cdiv(L_KEEP * D_MODEL, 256), 256, 0, stream>>>(fbuf, f1, L_KEEP * D_MODEL);
    run_mamba(xln, m2f, mo, L_KEEP);
    k_add_rows<<<cdiv(L_KEEP * D_MODEL, 256), 256, 0, stream>>>(f1, mo, L_KEEP, 0);
    run_mamba(xlnf, m2b, mo, L_KEEP);
    k_add_rows<<<cdiv(L_KEEP * D_MODEL, 256), 256, 0, stream>>>(f1, mo, L_KEEP, 1);

    // -------- Pooling head: out = softmax(attn(LN(f1))) . LN(f1) --------
    k_layernorm<<<L_KEEP, 256, 0, stream>>>(f1, norm_w, norm_b, hn);   // r2 in hn
    gemm(hn, D_MODEL, grp_w1_bf, t1, L_KEEP, 128, D_MODEL);
    k_tanh_w2<<<L_KEEP, 128, 0, stream>>>(t1, grp_b1, grp_w2, grp_b2, sc);
    k_softmax_1b<<<1, 256, 0, stream>>>(sc, L_KEEP);
    k_wsum<<<cdiv(D_MODEL, 256), 256, 0, stream>>>(hn, sc, (float*)d_out, L_KEEP);
}
